// newModel_65420941853225
// MI455X (gfx1250) — matrix operand order repaired and checked by reading, not compiled, not run on hardware
//
#include <hip/hip_runtime.h>
#include <hip/hip_bf16.h>

#define N_NODES 50000
#define N_EDGES 1600000
#define G_EMB   128
#define L_EMB   256
#define HID     256
#define SEQ     2048
#define TLEN    16
#define T_STEPS (SEQ - TLEN)        /* 2032 */
#define GW      (4 * HID)           /* 1024 gate width */

typedef __bf16 bf16_t;
typedef __attribute__((ext_vector_type(16))) __bf16 v16bf;
typedef __attribute__((ext_vector_type(8)))  float  v8f;

__device__ __forceinline__ bf16_t f2bf(float f) {
    unsigned u = __float_as_uint(f);
    unsigned r = (u + 0x7FFFu + ((u >> 16) & 1u)) >> 16;   // RNE
    unsigned short s = (unsigned short)r;
    return __builtin_bit_cast(bf16_t, s);
}

__device__ __forceinline__ float sigmoidf_(float x) {
    return 1.0f / (1.0f + __expf(-x));
}

// ---------------------------------------------------------------------------
// GCN: support = emb_loc_graph[loc_train] @ gc_w   (50000x128 @ 128x256)
// One wave -> one 16x16 output tile, K-loop of 128/32 = 4 WMMAs.
// ---------------------------------------------------------------------------
__global__ void gcn_support_kernel(const int* __restrict__ loc_train,
                                   const float* __restrict__ emb_g,
                                   const float* __restrict__ gc_w,
                                   float* __restrict__ support) {
    const int mt   = blockIdx.x;                         // 0..3124
    const int nt   = blockIdx.y * 8 + threadIdx.y;       // 0..15
    const int lane = threadIdx.x;                        // 0..31
    const int half = lane >> 4;
    const int mn   = lane & 15;
    const int n0   = nt * 16;

    const int node = loc_train[mt * 16 + mn];
    const float* arow = emb_g + (size_t)node * G_EMB;

    v8f acc = {};
    for (int k0 = 0; k0 < G_EMB; k0 += 32) {
        v16bf a, b;
#pragma unroll
        for (int e = 0; e < 16; ++e) {
            const int ka = ((e < 8) ? e : e + 8) + 8 * half;
            a[e] = f2bf(arow[k0 + ka]);
            const int kb = ((e < 8) ? e : e + 8) + 8 * half;
            b[e] = f2bf(gc_w[(size_t)(k0 + kb) * HID + (n0 + mn)]);
        }
        acc = __builtin_amdgcn_wmma_f32_16x16x32_bf16(false, a, false, b,
                                                      (short)0, acc, false, false);
    }
#pragma unroll
    for (int r = 0; r < 8; ++r) {
        const int m = mt * 16 + r + 8 * half;
        support[(size_t)m * HID + (n0 + mn)] = acc[r];
    }
}

// ---------------------------------------------------------------------------
// zero the aggregation buffer (must be re-done every launch: atomics below)
// ---------------------------------------------------------------------------
__global__ void zero_kernel(float* __restrict__ p, size_t n) {
    size_t i = (size_t)blockIdx.x * blockDim.x + threadIdx.x;
    size_t stride = (size_t)gridDim.x * blockDim.x;
    for (; i < n; i += stride) p[i] = 0.0f;
}

// ---------------------------------------------------------------------------
// Edge scatter: agg[row] += w * support[col]  (4 channels / thread)
// ---------------------------------------------------------------------------
__global__ void gcn_edge_kernel(const int* __restrict__ edge_index,
                                const float* __restrict__ edge_weight,
                                const float* __restrict__ support,
                                float* __restrict__ agg) {
    const size_t gid = (size_t)blockIdx.x * blockDim.x + threadIdx.x;
    if (gid >= (size_t)N_EDGES * (HID / 4)) return;
    const int e  = (int)(gid >> 6);            // HID/4 == 64 channel-groups
    const int c4 = (int)(gid & 63) * 4;
    const int row = edge_index[e];
    const int col = edge_index[N_EDGES + e];
    const float w = edge_weight[e];
    const float4 s = *(const float4*)(support + (size_t)col * HID + c4);
    float* dst = agg + (size_t)row * HID + c4;
    atomicAdd(dst + 0, w * s.x);
    atomicAdd(dst + 1, w * s.y);
    atomicAdd(dst + 2, w * s.z);
    atomicAdd(dst + 3, w * s.w);
}

// ---------------------------------------------------------------------------
// c1_gcn[c] = mean_n relu(agg[n][c] + gc_b[c]); one block per channel
// ---------------------------------------------------------------------------
__global__ void gcn_mean_kernel(const float* __restrict__ agg,
                                const float* __restrict__ gc_b,
                                float* __restrict__ out_c1gcn) {
    const int c = blockIdx.x;
    const float b = gc_b[c];
    float part = 0.0f;
    for (int n = threadIdx.x; n < N_NODES; n += blockDim.x) {
        const float v = agg[(size_t)n * HID + c] + b;
        part += (v > 0.0f) ? v : 0.0f;
    }
    __shared__ float red[256];
    red[threadIdx.x] = part;
    __syncthreads();
    for (int s = 128; s > 0; s >>= 1) {
        if (threadIdx.x < s) red[threadIdx.x] += red[threadIdx.x + s];
        __syncthreads();
    }
    if (threadIdx.x == 0) out_c1gcn[c] = red[0] / (float)N_NODES;
}

// ---------------------------------------------------------------------------
// LSTM input projection: gates[t][:] = emb_loc[loc[t]] @ Wih^T + bih + bhh
// (2032x256 @ 256x1024) via WMMA; one wave -> 16x16 tile, K-loop 256/32 = 8.
// ---------------------------------------------------------------------------
__global__ void lstm_xw_kernel(const int* __restrict__ loc,
                               const float* __restrict__ emb_loc,
                               const float* __restrict__ Wih,  // (1024,256) row-major
                               const float* __restrict__ bih,
                               const float* __restrict__ bhh,
                               float* __restrict__ gates) {
    const int mt   = blockIdx.x;                         // 0..126
    const int nt   = blockIdx.y * 8 + threadIdx.y;       // 0..63
    const int lane = threadIdx.x;
    const int half = lane >> 4;
    const int mn   = lane & 15;
    const int n0   = nt * 16;

    const int tok = loc[mt * 16 + mn];
    const float* xrow = emb_loc + (size_t)tok * L_EMB;
    const float* wrow = Wih + (size_t)(n0 + mn) * L_EMB;  // B[k][n] = Wih[n][k]

    v8f acc = {};
    for (int k0 = 0; k0 < L_EMB; k0 += 32) {
        v16bf a, b;
#pragma unroll
        for (int e = 0; e < 16; ++e) {
            const int ka = ((e < 8) ? e : e + 8) + 8 * half;
            a[e] = f2bf(xrow[k0 + ka]);
            const int kb = ((e < 8) ? e : e + 8) + 8 * half;
            b[e] = f2bf(wrow[k0 + kb]);
        }
        acc = __builtin_amdgcn_wmma_f32_16x16x32_bf16(false, a, false, b,
                                                      (short)0, acc, false, false);
    }
#pragma unroll
    for (int r = 0; r < 8; ++r) {
        const int t   = mt * 16 + r + 8 * half;
        const int col = n0 + mn;
        gates[(size_t)t * GW + col] = acc[r] + bih[col] + bhh[col];
    }
}

// ---------------------------------------------------------------------------
// Recurrence: one block per direction, 1024 threads (one per gate row).
// h, c, gate pre-activations live in LDS; 2032 sequential steps.
// Writes hidden_history + hT/cT directly into d_out.
// ---------------------------------------------------------------------------
__global__ void lstm_rec_kernel(const float* __restrict__ gates_f,
                                const float* __restrict__ gates_b,
                                const float* __restrict__ Whh_f,
                                const float* __restrict__ Whh_b,
                                float* __restrict__ out) {
    const int dir = blockIdx.x;   // 0 = forward, 1 = backward
    const int g   = threadIdx.x;  // 0..1023
    const float* gates = dir ? gates_b : gates_f;
    const float* Whh   = dir ? Whh_b   : Whh_f;

    __shared__ float h[HID];
    __shared__ float c[HID];
    __shared__ float gs[GW];
    if (g < HID) { h[g] = 0.0f; c[g] = 0.0f; }
    __syncthreads();

    const float4* wrow = (const float4*)(Whh + (size_t)g * HID);

    for (int s = 0; s < T_STEPS; ++s) {
        const int t = dir ? (T_STEPS - 1 - s) : s;
        // prefetch next step's input-projection row (global_prefetch_b8)
        if (s + 1 < T_STEPS) {
            const int tn = dir ? (t - 1) : (t + 1);
            __builtin_prefetch(gates + (size_t)tn * GW + g, 0, 3);
        }
        float acc = gates[(size_t)t * GW + g];
#pragma unroll 8
        for (int k = 0; k < HID / 4; ++k) {
            const float4 w = wrow[k];
            acc += w.x * h[4 * k + 0] + w.y * h[4 * k + 1]
                 + w.z * h[4 * k + 2] + w.w * h[4 * k + 3];
        }
        gs[g] = acc;
        __syncthreads();
        if (g < HID) {
            const float ig = sigmoidf_(gs[g]);
            const float fg = sigmoidf_(gs[HID + g]);
            const float gg = tanhf(gs[2 * HID + g]);
            const float og = sigmoidf_(gs[3 * HID + g]);
            const float cn = fg * c[g] + ig * gg;
            const float hn = og * tanhf(cn);
            c[g] = cn;
            h[g] = hn;
            out[(size_t)t * (2 * HID) + dir * HID + g] = hn;  // hidden_history
        }
        __syncthreads();
    }
    if (g < HID) {
        const size_t HH = (size_t)T_STEPS * (2 * HID);
        out[HH + HID + (size_t)dir * HID + g]            = h[g];  // h1
        out[HH + HID + 2 * HID + (size_t)dir * HID + g]  = c[g];  // c1
    }
}

// ---------------------------------------------------------------------------
extern "C" void kernel_launch(void* const* d_in, const int* in_sizes, int n_in,
                              void* d_out, int out_size, void* d_ws, size_t ws_size,
                              hipStream_t stream) {
    const int*   loc        = (const int*)  d_in[0];
    const int*   loc_train  = (const int*)  d_in[1];
    const int*   edge_index = (const int*)  d_in[2];
    const float* edge_w     = (const float*)d_in[3];
    /* d_in[4] = target_len (hardcoded 16) */
    const float* emb_loc    = (const float*)d_in[5];
    const float* emb_g      = (const float*)d_in[6];
    const float* gc_w       = (const float*)d_in[7];
    const float* gc_b       = (const float*)d_in[8];
    const float* Wih_f      = (const float*)d_in[9];
    const float* Whh_f      = (const float*)d_in[10];
    const float* bih_f      = (const float*)d_in[11];
    const float* bhh_f      = (const float*)d_in[12];
    const float* Wih_b      = (const float*)d_in[13];
    const float* Whh_b      = (const float*)d_in[14];
    const float* bih_b      = (const float*)d_in[15];
    const float* bhh_b      = (const float*)d_in[16];
    float* out = (float*)d_out;

    // workspace layout (floats)
    float* support = (float*)d_ws;                               // 50000*256
    float* agg     = support + (size_t)N_NODES * HID;            // 50000*256
    float* gates_f = agg + (size_t)N_NODES * HID;                // 2032*1024
    float* gates_b = gates_f + (size_t)T_STEPS * GW;             // 2032*1024

    const size_t HH = (size_t)T_STEPS * (2 * HID);

    // ---- GCN ----
    zero_kernel<<<2048, 256, 0, stream>>>(agg, (size_t)N_NODES * HID);
    gcn_support_kernel<<<dim3(N_NODES / 16, 2), dim3(32, 8), 0, stream>>>(
        loc_train, emb_g, gc_w, support);
    {
        const size_t work = (size_t)N_EDGES * (HID / 4);
        gcn_edge_kernel<<<(unsigned)((work + 255) / 256), 256, 0, stream>>>(
            edge_index, edge_w, support, agg);
    }
    gcn_mean_kernel<<<HID, 256, 0, stream>>>(agg, gc_b, out + HH);

    // ---- LSTM input projections (WMMA GEMMs) ----
    lstm_xw_kernel<<<dim3(T_STEPS / 16, 8), dim3(32, 8), 0, stream>>>(
        loc, emb_loc, Wih_f, bih_f, bhh_f, gates_f);
    lstm_xw_kernel<<<dim3(T_STEPS / 16, 8), dim3(32, 8), 0, stream>>>(
        loc, emb_loc, Wih_b, bih_b, bhh_b, gates_b);

    // ---- Recurrence (both directions in parallel) ----
    lstm_rec_kernel<<<2, GW, 0, stream>>>(gates_f, gates_b, Whh_f, Whh_b, out);
}
